// GraphAttentionLayerSim_249108103458
// MI455X (gfx1250) — matrix operand order repaired and checked by reading, not compiled, not run on hardware
//
#include <hip/hip_runtime.h>
#include <hip/hip_bf16.h>
#include <math.h>

typedef __attribute__((ext_vector_type(16))) _Float16 v16h;
typedef __attribute__((ext_vector_type(8)))  _Float16 v8h;
typedef __attribute__((ext_vector_type(8)))  float    v8f;
typedef __attribute__((ext_vector_type(8)))  int      v8i;
typedef __attribute__((ext_vector_type(4)))  int      v4i;
typedef __attribute__((ext_vector_type(4)))  unsigned int u32x4;
typedef __attribute__((ext_vector_type(8)))  int      i32x8;

#define BATCH 16
#define NN 2048
#define FF 128
#define NEG_BIG (-9.0e15f)

// LDS tile geometry (halfs). Key tile staged per 32-key chunk, two layouts:
//   n-major [32][128] padded to row stride 136  -> A fragments
//   f-major [128][32] padded to row stride 40   -> B fragments of P@V
#define XH_STRIDE 136
#define XT_STRIDE 40
#define XT_OFF    (32 * XH_STRIDE)              // 4352 halfs
#define BUF_H     (XT_OFF + FF * XT_STRIDE)     // 9472 halfs per buffer

#if __has_builtin(__builtin_amdgcn_tensor_load_to_lds) && __has_builtin(__builtin_amdgcn_s_wait_tensorcnt)
#define USE_TDM 1
#else
#define USE_TDM 0
#endif

// ---------------------------------------------------------------------------
// Prep kernel 1: f32 -> f16 copy of x
// ---------------------------------------------------------------------------
__global__ void cvt_x_f16(const float* __restrict__ x, _Float16* __restrict__ xh, int n) {
    int i = blockIdx.x * blockDim.x + threadIdx.x;
    if (i < n) xh[i] = (_Float16)x[i];
}

// ---------------------------------------------------------------------------
// Prep kernel 2: W_a (f,g) -> Wt f16 (g,f)
// ---------------------------------------------------------------------------
__global__ void cvt_w_t(const float* __restrict__ W, _Float16* __restrict__ wt) {
    int i = blockIdx.x * blockDim.x + threadIdx.x;   // i = g*128 + f
    if (i < FF * FF) {
        int g = i >> 7, f = i & 127;
        wt[i] = (_Float16)W[f * FF + g];
    }
}

// ---------------------------------------------------------------------------
// Prep kernel 3: tiled transpose  x[b][n][f] -> xth[b][f][n]  (f16)
// ---------------------------------------------------------------------------
__global__ void transpose_x(const float* __restrict__ x, _Float16* __restrict__ xth) {
    __shared__ float tile[32][33];
    int bid = blockIdx.x;
    int ft = bid % (FF / 32); bid /= (FF / 32);
    int nt = bid % (NN / 32); int b = bid / (NN / 32);
    int n0 = nt * 32, f0 = ft * 32;
    int tx = threadIdx.x, ty = threadIdx.y;
    #pragma unroll
    for (int j = 0; j < 4; ++j)
        tile[ty + 8 * j][tx] = x[((size_t)b * NN + n0 + ty + 8 * j) * FF + f0 + tx];
    __syncthreads();
    #pragma unroll
    for (int j = 0; j < 4; ++j)
        xth[((size_t)b * FF + f0 + ty + 8 * j) * NN + n0 + tx] = (_Float16)tile[tx][ty + 8 * j];
}

// ---------------------------------------------------------------------------
// Prep kernel 4: xw = x @ W_a via WMMA. Loads batched per K-chunk so the 8
// WMMAs issue back-to-back behind a single wait.
// ---------------------------------------------------------------------------
__global__ __launch_bounds__(32) void gemm_xw(const _Float16* __restrict__ xh,
                                              const _Float16* __restrict__ wt,
                                              _Float16* __restrict__ xwh) {
    const int lane = threadIdx.x;
    const int lh = lane & 15, hh = lane >> 4;
    const int tile = blockIdx.x;
    const int b = tile >> 7;
    const int n_base = (tile & 127) << 4;

    v16h A[4];
    const _Float16* xr = xh + ((size_t)b * NN + n_base + lh) * FF;
    #pragma unroll
    for (int kc = 0; kc < 4; ++kc) {
        v8h lo = *(const v8h*)(xr + 32 * kc + 16 * hh);
        v8h hi = *(const v8h*)(xr + 32 * kc + 16 * hh + 8);
        #pragma unroll
        for (int i = 0; i < 8; ++i) { A[kc][i] = lo[i]; A[kc][8 + i] = hi[i]; }
    }

    v8f c[8];
    #pragma unroll
    for (int gt = 0; gt < 8; ++gt) c[gt] = (v8f){};

    #pragma unroll
    for (int kc = 0; kc < 4; ++kc) {
        v16h Bt[8];
        #pragma unroll
        for (int gt = 0; gt < 8; ++gt)
            Bt[gt] = *(const v16h*)(wt + (size_t)(16 * gt + lh) * FF + 32 * kc + 16 * hh);
        #pragma unroll
        for (int gt = 0; gt < 8; ++gt)
            c[gt] = __builtin_amdgcn_wmma_f32_16x16x32_f16(false, A[kc], false, Bt[gt],
                                                           (short)0, c[gt], false, false);
    }

    #pragma unroll
    for (int gt = 0; gt < 8; ++gt)
        #pragma unroll
        for (int r = 0; r < 8; ++r)
            xwh[((size_t)b * NN + n_base + r + 8 * hh) * FF + 16 * gt + lh] = (_Float16)c[gt][r];
}

// ---------------------------------------------------------------------------
// TDM helper: 2D tensor tile -> LDS, element size 2B, with LDS row padding.
// Descriptor bit layout per CDNA5 ISA section 8.3 / 8.4.
//   pad_interval code: pad after 2^(code+1) DWORDs ; pad_amount code: code+1 DWORDs
// NOTE: this toolchain exposes the 6-arg (clang-23 / therock) form of
// __builtin_amdgcn_tensor_load_to_lds: (u32x4, i32x8, i32x4, i32x4, i32x8, i32).
// ---------------------------------------------------------------------------
#if USE_TDM
__device__ __forceinline__ void tdm_load_2d(const void* gaddr, unsigned lds_addr,
                                            int tile_d0, int tile_d1,
                                            int tens_d0, int tens_d1,
                                            int stride0,
                                            int pad_interval, int pad_amount) {
    unsigned long long ga = (unsigned long long)(size_t)gaddr;
    u32x4 g0;
    g0[0] = 1u;                                        // count=1, load, no gather
    g0[1] = lds_addr;                                  // LDS byte address
    g0[2] = (unsigned)(ga & 0xFFFFFFFFu);              // global_addr[31:0]
    g0[3] = (unsigned)((ga >> 32) & 0x1FFFFFFu)        // global_addr[56:32]
          | (2u << 30);                                // type = 2 (image)
    i32x8 g1;
    g1[0] = (1 << 16)                                  // data_size = 2 bytes
          | (1 << 20)                                  // pad_enable
          | (pad_interval << 22)
          | (pad_amount << 25);
    g1[1] = (tens_d0 & 0xFFFF) << 16;                  // tensor_dim0[15:0]
    g1[2] = ((unsigned)tens_d0 >> 16) | ((tens_d1 & 0xFFFF) << 16);
    g1[3] = ((unsigned)tens_d1 >> 16) | (tile_d0 << 16);
    g1[4] = tile_d1 & 0xFFFF;                          // tile_dim2 = 0 (2D)
    g1[5] = stride0;                                   // tensor_dim0_stride[31:0]
    g1[6] = 0;
    g1[7] = 0;
    v4i  z4 = (v4i){0, 0, 0, 0};                       // groups 2/3: unused (2D)
    i32x8 z8 = (i32x8){0, 0, 0, 0, 0, 0, 0, 0};
    __builtin_amdgcn_tensor_load_to_lds(g0, g1, z4, z4, z8, 0);
}
#endif

// ---------------------------------------------------------------------------
// Main fused kernel. 4 waves/WG, 16 query rows each. Key tile (both layouts)
// double-buffered through LDS via the Tensor Data Mover (TENSORcnt pipeline).
// S computed transposed so softmax P is already the A-fragment of P@V.
// ---------------------------------------------------------------------------
__global__ __launch_bounds__(128) void gat_main(const int*   __restrict__ adj,
                                                const float* __restrict__ bias,
                                                const _Float16* __restrict__ xh,
                                                const _Float16* __restrict__ xth,
                                                const _Float16* __restrict__ xwh,
                                                float* __restrict__ out) {
    __shared__ __align__(32) _Float16 sm[2][BUF_H];

    const int t    = threadIdx.x;        // 0..127
    const int lane = t & 31;
    const int wv   = t >> 5;             // wave id 0..3
    const int lh = lane & 15, hh = lane >> 4;
    const int b      = blockIdx.x >> 5;                 // 32 WGs per batch
    const int n_base = ((blockIdx.x & 31) << 6) + (wv << 4);

    const _Float16* xb  = xh  + (size_t)b * NN * FF;
    const _Float16* xtb = xth + (size_t)b * FF * NN;
    const int*      adjr = adj + ((size_t)b * NN + n_base + lh) * NN;

#if USE_TDM
    auto stage_issue = [&](int mb, int buf) {
        if (wv == 0) {   // one wave drives the TDM; wave-uniform branch
            unsigned lbase = (unsigned)(size_t)(&sm[buf][0]);
            // n-major tile: rows of 128 halfs, pad 4 DW after every 64 DW -> stride 136
            tdm_load_2d(xb + (size_t)mb * FF, lbase,
                        FF, 32, FF, NN, FF, /*interval=*/5, /*amount=*/3);
            // f-major tile: rows of 32 halfs, pad 4 DW after every 16 DW -> stride 40
            tdm_load_2d(xtb + mb, lbase + XT_OFF * 2,
                        32, FF, NN, FF, NN, /*interval=*/3, /*amount=*/3);
        }
    };
    auto stage_wait = [&]() {
        if (wv == 0) __builtin_amdgcn_s_wait_tensorcnt(0);
    };
#else
    v4i sA[4], sB[4];
    auto stage_load = [&](int mb) {
        const v4i* gA = (const v4i*)(xb + (size_t)mb * FF);   // contiguous 8KB
        #pragma unroll
        for (int j = 0; j < 4; ++j) sA[j] = gA[t + 128 * j];
        #pragma unroll
        for (int j = 0; j < 4; ++j) {
            int p = t + 128 * j;
            sB[j] = *(const v4i*)(xtb + (size_t)(p >> 2) * NN + mb + 8 * (p & 3));
        }
    };
    auto stage_store = [&](int buf) {
        _Float16* d = &sm[buf][0];
        #pragma unroll
        for (int j = 0; j < 4; ++j) {
            int p = t + 128 * j;
            *(v4i*)(d + (p >> 4) * XH_STRIDE + (p & 15) * 8) = sA[j];
        }
        #pragma unroll
        for (int j = 0; j < 4; ++j) {
            int p = t + 128 * j;
            *(v4i*)(d + XT_OFF + (p >> 2) * XT_STRIDE + (p & 3) * 8) = sB[j];
        }
    };
#endif

    // B operand of S^T: xw rows n (fixed for wave) -> preload 4 K-chunks
    v16h Bw[4];
    {
        const _Float16* xwr = xwh + ((size_t)b * NN + n_base + lh) * FF;
        #pragma unroll
        for (int kc = 0; kc < 4; ++kc)
            Bw[kc] = *(const v16h*)(xwr + 32 * kc + 16 * hh);
    }

    v8f acc[8];
    #pragma unroll
    for (int ft = 0; ft < 8; ++ft) acc[ft] = (v8f){};
    float rowM = -INFINITY, rowL = 0.0f;

    // prologue: stage chunk 0 into buffer 0
#if USE_TDM
    stage_issue(0, 0);
    stage_wait();
#else
    stage_load(0);
    stage_store(0);
#endif
    __syncthreads();
    int cur = 0;

    for (int mb = 0; mb < NN; mb += 32) {
        const int mb_next = (mb + 32) & (NN - 1);
#if USE_TDM
        stage_issue(mb_next, cur ^ 1);   // TDM streams while we compute below
#else
        stage_load(mb_next);
#endif
        v8i ad0 = *(const v8i*)(adjr + mb + 8 * hh);
        v8i ad1 = *(const v8i*)(adjr + mb + 16 + 8 * hh);
        __builtin_prefetch(adjr + mb_next + 8 * hh, 0, 0);       // global_prefetch_b8

        const _Float16* L = &sm[cur][0];

        // ---- S^T = keys x queries (16 keys per subtile) ----
        v8f S0 = (v8f){}, S1 = (v8f){};
        #pragma unroll
        for (int kc = 0; kc < 4; ++kc) {
            const _Float16* a0p = L + lh * XH_STRIDE + 32 * kc + 16 * hh;
            const _Float16* a1p = L + (16 + lh) * XH_STRIDE + 32 * kc + 16 * hh;
            v8h l0 = *(const v8h*)(a0p);
            v8h h0 = *(const v8h*)(a0p + 8);
            v8h l1 = *(const v8h*)(a1p);
            v8h h1 = *(const v8h*)(a1p + 8);
            v16h A0, A1;
            #pragma unroll
            for (int i = 0; i < 8; ++i) {
                A0[i] = l0[i]; A0[8 + i] = h0[i];
                A1[i] = l1[i]; A1[8 + i] = h1[i];
            }
            S0 = __builtin_amdgcn_wmma_f32_16x16x32_f16(false, A0, false, Bw[kc],
                                                        (short)0, S0, false, false);
            S1 = __builtin_amdgcn_wmma_f32_16x16x32_f16(false, A1, false, Bw[kc],
                                                        (short)0, S1, false, false);
        }

        // ---- leaky relu + adjacency mask; S0[r] is key m = mb + 8*hh + r ----
        float cm = -INFINITY;
        #pragma unroll
        for (int r = 0; r < 8; ++r) {
            float s = S0[r]; s = fmaxf(s, 0.1f * s); s = (ad0[r] > 0) ? s : NEG_BIG;
            float tt = S1[r]; tt = fmaxf(tt, 0.1f * tt); tt = (ad1[r] > 0) ? tt : NEG_BIG;
            S0[r] = s; S1[r] = tt;
            cm = fmaxf(cm, fmaxf(s, tt));
        }
        cm = fmaxf(cm, __shfl_xor(cm, 16, 32));

        // ---- online softmax; rescale only when the max actually moved ----
        float Mnew = fmaxf(rowM, cm);
        if (__ballot(cm > rowM)) {           // wave-uniform branch, EXEC untouched
            float scale = __expf(rowM - Mnew);
            rowL *= scale;
            float sc[8];
            #pragma unroll
            for (int r = 0; r < 8; ++r) sc[r] = __shfl(scale, r + 8 * hh, 32);
            #pragma unroll
            for (int ft = 0; ft < 8; ++ft)
                #pragma unroll
                for (int r = 0; r < 8; ++r) acc[ft][r] *= sc[r];
        }
        rowM = Mnew;

        float csum = 0.0f;
        v16h Pa;                              // A-fragment of P, no shuffles needed
        #pragma unroll
        for (int r = 0; r < 8; ++r) {
            float p = __expf(S0[r] - Mnew); csum += p; Pa[r]     = (_Float16)p;
            float q = __expf(S1[r] - Mnew); csum += q; Pa[8 + r] = (_Float16)q;
        }
        csum += __shfl_xor(csum, 16, 32);
        rowL += csum;

        // ---- P @ V : 8 f-tiles, B fragments from f-major LDS tile ----
        #pragma unroll
        for (int ft = 0; ft < 8; ++ft) {
            const _Float16* bp = L + XT_OFF + (16 * ft + lh) * XT_STRIDE + 8 * hh;
            v8h lo = *(const v8h*)(bp);
            v8h hi = *(const v8h*)(bp + 16);
            v16h Bx;
            #pragma unroll
            for (int i = 0; i < 8; ++i) { Bx[i] = lo[i]; Bx[8 + i] = hi[i]; }
            acc[ft] = __builtin_amdgcn_wmma_f32_16x16x32_f16(false, Pa, false, Bx,
                                                             (short)0, acc[ft], false, false);
        }

        // publish next chunk, one barrier per iteration
#if USE_TDM
        stage_wait();
#else
        stage_store(cur ^ 1);
#endif
        __syncthreads();
        cur ^= 1;
    }

    // ---- finalize: /L, +bias, ELU, store ----
    float Lr[8];
    #pragma unroll
    for (int r = 0; r < 8; ++r) Lr[r] = __shfl(rowL, r + 8 * hh, 32);

    #pragma unroll
    for (int ft = 0; ft < 8; ++ft) {
        float bv = bias[16 * ft + lh];
        #pragma unroll
        for (int r = 0; r < 8; ++r) {
            float v = acc[ft][r] / Lr[r] + bv;
            v = (v > 0.0f) ? v : (__expf(v) - 1.0f);
            out[((size_t)b * NN + n_base + r + 8 * hh) * FF + 16 * ft + lh] = v;
        }
    }
}

// ---------------------------------------------------------------------------
extern "C" void kernel_launch(void* const* d_in, const int* in_sizes, int n_in,
                              void* d_out, int out_size, void* d_ws, size_t ws_size,
                              hipStream_t stream) {
    const float* x    = (const float*)d_in[0];   // [B,N,F]
    const int*   adj  = (const int*)d_in[1];     // [B,N,N]
    const float* W    = (const float*)d_in[2];   // [F,F]
    const float* bias = (const float*)d_in[3];   // [1,F]
    float* out = (float*)d_out;

    char* ws = (char*)d_ws;
    const size_t sz = (size_t)BATCH * NN * FF;
    _Float16* xh  = (_Float16*)(ws);                 // 8 MB
    _Float16* xth = (_Float16*)(ws + sz * 2);        // 8 MB
    _Float16* xwh = (_Float16*)(ws + sz * 4);        // 8 MB
    _Float16* wt  = (_Float16*)(ws + sz * 6);        // 32 KB

    int total = (int)sz;
    cvt_x_f16<<<(total + 255) / 256, 256, 0, stream>>>(x, xh, total);
    cvt_w_t<<<(FF * FF + 255) / 256, 256, 0, stream>>>(W, wt);
    transpose_x<<<BATCH * (NN / 32) * (FF / 32), dim3(32, 8), 0, stream>>>(x, xth);
    gemm_xw<<<BATCH * (NN / 16), 32, 0, stream>>>(xh, wt, xwh);
    gat_main<<<BATCH * (NN / 64), 128, 0, stream>>>(adj, bias, xh, xth, xwh, out);
}
